// MatchPredictor_21440476741640
// MI455X (gfx1250) — matrix operand order repaired and checked by reading, not compiled, not run on hardware
//
#include <hip/hip_runtime.h>
#include <hip/hip_bf16.h>
#include <math.h>

#define BROWS 32768
#define TM 16

typedef __attribute__((ext_vector_type(16))) _Float16 v16h;
typedef __attribute__((ext_vector_type(8)))  float    v8f;

// ---- packed-weight fragment layout -----------------------------------------
// Each B-fragment (one 32x16 WMMA tile) = 32 lanes x 16 halves, contiguous,
// 1 KB per tile, 32-byte aligned per lane. Tile index = base + kt*NT + nt.
#define TB_HOME  0    // KT=1 NT=4
#define TB_AWAY  4    // KT=1 NT=4
#define TB_MC    8    // KT=1 NT=4
#define TB_WIN   12   // KT=2 NT=16
#define TB_XPROJ 44   // KT=4 NT=3
#define TB_MOUT  56   // KT=4 NT=4
#define TB_C1    72   // KT=6 NT=4
#define N_TILES  96
#define PACK_HALVES ((size_t)N_TILES * 32 * 16)   // 49152 halves = 96 KB

__device__ __forceinline__ int laneid() { return threadIdx.x & 31; }

__device__ __forceinline__ v8f zero8() {
  v8f z = {0.f, 0.f, 0.f, 0.f, 0.f, 0.f, 0.f, 0.f};
  return z;
}

// A-fragment from LDS f16 row-major matrix (contiguous k = 16*(lane/16) + i, matches B).
__device__ __forceinline__ v16h ldsA(const _Float16* base, int ld, int k0) {
  const int lane = laneid();
  const int m = lane & 15;
  const _Float16* p = base + m * ld + k0 + ((lane >> 4) << 4);
  v16h a;
#pragma unroll
  for (int i = 0; i < 8; ++i) a[i] = p[i];
#pragma unroll
  for (int i = 0; i < 8; ++i) a[i + 8] = p[8 + i];
  return a;
}

// Fallback: build B-fragment from global f32 weights with zero padding.
__device__ __forceinline__ v16h gB(const float* __restrict__ W, int ldn,
                                   int k0, int n0, int krows, int ncols) {
  const int lane = laneid();
  const int n = n0 + (lane & 15);
  const int kb = k0 + ((lane >> 4) << 4);
  const bool nok = (n < ncols);
  v16h b;
#pragma unroll
  for (int i = 0; i < 16; ++i) {
    const int k = kb + i;
    const float v = (nok && (k < krows)) ? W[(size_t)k * ldn + n] : 0.0f;
    b[i] = (_Float16)v;
  }
  return b;
}

// Fast path: one aligned 32-byte vector load (2x global_load_b128), coalesced.
__device__ __forceinline__ v16h pB(const _Float16* __restrict__ wpk, int tile) {
  const _Float16* p = wpk + (((size_t)tile << 5) + laneid()) * 16;
  return *reinterpret_cast<const v16h*>(p);
}

__device__ __forceinline__ v8f wmma_f16(v16h a, v16h b, v8f c) {
  return __builtin_amdgcn_wmma_f32_16x16x32_f16(false, a, false, b, (short)0,
                                                c, false, false);
}

__device__ __forceinline__ float siluf(float v) {
  return v / (1.0f + __expf(-v));
}

// ---- one-shot weight pack kernel (f32 -> f16 fragment-major, zero padded) ---
__global__ __launch_bounds__(256) void pack_weights(
    const float* __restrict__ W_home, const float* __restrict__ W_away,
    const float* __restrict__ W_mc, const float* __restrict__ W_in,
    const float* __restrict__ W_xproj, const float* __restrict__ W_mout,
    const float* __restrict__ W_c1, _Float16* __restrict__ wpk) {
  const int t = blockIdx.x * 256 + threadIdx.x;
  if (t >= (int)PACK_HALVES) return;
  const int tile = t >> 9;
  const int lane = (t >> 4) & 31;
  const int i = t & 15;
  const float* W;
  int ld, kr, nc, NT, l;
  if (tile < TB_AWAY)       { W = W_home;  ld = 64;  kr = 22;  nc = 64;  NT = 4;  l = tile; }
  else if (tile < TB_MC)    { W = W_away;  ld = 64;  kr = 22;  nc = 64;  NT = 4;  l = tile - TB_AWAY; }
  else if (tile < TB_WIN)   { W = W_mc;    ld = 64;  kr = 20;  nc = 64;  NT = 4;  l = tile - TB_MC; }
  else if (tile < TB_XPROJ) { W = W_in;    ld = 256; kr = 64;  nc = 256; NT = 16; l = tile - TB_WIN; }
  else if (tile < TB_MOUT)  { W = W_xproj; ld = 36;  kr = 128; nc = 36;  NT = 3;  l = tile - TB_XPROJ; }
  else if (tile < TB_C1)    { W = W_mout;  ld = 64;  kr = 128; nc = 64;  NT = 4;  l = tile - TB_MOUT; }
  else                      { W = W_c1;    ld = 64;  kr = 192; nc = 64;  NT = 4;  l = tile - TB_C1; }
  const int kt = l / NT, nt = l % NT;
  const int n = nt * 16 + (lane & 15);
  const int k = kt * 32 + ((lane >> 4) << 4) + i;
  const float v = (k < kr && n < nc) ? W[(size_t)k * ld + n] : 0.0f;
  wpk[t] = (_Float16)v;
}

// ---- fused model kernel: one wave per block, 16 rows ------------------------
template <bool PACKED>
__global__ __launch_bounds__(32) void match_predictor_fused(
    const float* __restrict__ x,
    const float* __restrict__ W_home, const float* __restrict__ b_home,
    const float* __restrict__ W_away, const float* __restrict__ b_away,
    const float* __restrict__ W_in,
    const float* __restrict__ conv_w, const float* __restrict__ conv_b,
    const float* __restrict__ W_xproj,
    const float* __restrict__ W_dt, const float* __restrict__ b_dt,
    const float* __restrict__ A_log, const float* __restrict__ D_skip,
    const float* __restrict__ W_mout,
    const float* __restrict__ W_mc, const float* __restrict__ b_mc,
    const float* __restrict__ bn1_g, const float* __restrict__ bn1_b,
    const float* __restrict__ bn1_m, const float* __restrict__ bn1_v,
    const float* __restrict__ W_c1, const float* __restrict__ b_c1,
    const float* __restrict__ bn2_g, const float* __restrict__ bn2_b,
    const float* __restrict__ bn2_m, const float* __restrict__ bn2_v,
    const float* __restrict__ W_c2, const float* __restrict__ b_c2,
    const _Float16* __restrict__ wpk, float* __restrict__ out) {
  __shared__ _Float16 sStage[TM][96];   // x split into 3 zero-padded K=32 slabs
  __shared__ _Float16 sComb[TM][192];   // [home | away | ctx_f] -> A of W_c1
  __shared__ _Float16 sxc[2][TM][128];  // xz x-half, then silu(conv) in place
  __shared__ _Float16 sz[2][TM][128];   // silu(z)
  __shared__ _Float16 sdt[2][TM][128];  // softplus dt
  __shared__ _Float16 sG[TM][128];      // 0.5*(g0+g1) -> A of W_mout
  __shared__ float sXdbl[TM][48];       // x_dbl (N 36->48)
  __shared__ float sB0[TM][16];
  __shared__ float sP[TM][16];
  __shared__ float sS0[TM];
  __shared__ float sS1[TM];
  __shared__ float sH1[TM][64];
  __shared__ float sAe[128][16];        // -exp(A_log)
  __shared__ float sWdt[4][128];        // W_dt
  __shared__ float sCw2[128], sCw3[128], sCb[128], sDsk[128];

  const int lane = laneid();
  const int r0 = blockIdx.x * TM;

  // ---- Phase 0: zero stage + cache small per-channel tensors in LDS ----
  for (int t = lane; t < TM * 96; t += 32) (&sStage[0][0])[t] = (_Float16)0.0f;
  for (int t = lane; t < 128 * 16; t += 32)
    (&sAe[0][0])[t] = -__expf(A_log[t]);
  for (int t = lane; t < 4 * 128; t += 32) (&sWdt[0][0])[t] = W_dt[t];
  for (int t = lane; t < 128; t += 32) {
    sCw2[t] = conv_w[t * 4 + 2];
    sCw3[t] = conv_w[t * 4 + 3];
    sCb[t] = conv_b[t];
    sDsk[t] = D_skip[t];
  }
  __syncthreads();
  for (int t = lane; t < TM * 64; t += 32) {
    const int r = t >> 6, c = t & 63;
    const float v = __builtin_nontemporal_load(&x[(size_t)(r0 + r) * 64 + c]);
    int seg, lc;
    if (c < 22)      { seg = 0; lc = c; }
    else if (c < 44) { seg = 1; lc = c - 22; }
    else             { seg = 2; lc = c - 44; }
    sStage[r][seg * 32 + lc] = (_Float16)v;
  }
  __syncthreads();

  // ---- Phase 1: home / away embeds + ctx BN-ReLU branch ----
  {
    const v16h ah = ldsA(&sStage[0][0], 96, 0);
    const v16h aa = ldsA(&sStage[0][32], 96, 0);
    const v16h ac = ldsA(&sStage[0][64], 96, 0);
#pragma unroll
    for (int nt = 0; nt < 4; ++nt) {
      const int n0 = nt * 16;
      const int n = n0 + (lane & 15);
      const int mb = (lane >> 4) << 3;
      v8f dh = wmma_f16(ah, PACKED ? pB(wpk, TB_HOME + nt)
                                   : gB(W_home, 64, 0, n0, 22, 64), zero8());
      v8f da = wmma_f16(aa, PACKED ? pB(wpk, TB_AWAY + nt)
                                   : gB(W_away, 64, 0, n0, 22, 64), zero8());
      v8f dc = wmma_f16(ac, PACKED ? pB(wpk, TB_MC + nt)
                                   : gB(W_mc, 64, 0, n0, 20, 64), zero8());
      const float bh = b_home[n], ba = b_away[n], bc = b_mc[n];
      const float g1 = bn1_g[n], bb1 = bn1_b[n], m1 = bn1_m[n];
      const float iv1 = rsqrtf(bn1_v[n] + 1e-5f);
#pragma unroll
      for (int r = 0; r < 8; ++r) {
        sComb[mb + r][n] = (_Float16)(dh[r] + bh);
        sComb[mb + r][64 + n] = (_Float16)(da[r] + ba);
        float cv = (dc[r] + bc - m1) * iv1 * g1 + bb1;
        sComb[mb + r][128 + n] = (_Float16)(cv > 0.f ? cv : 0.f);
      }
    }
  }
  __syncthreads();

  // ---- Phase 2: xz = seq @ W_in (pos 0 = home, pos 1 = away) ----
#pragma unroll
  for (int pos = 0; pos < 2; ++pos) {
    const v16h a0 = ldsA(&sComb[0][pos * 64], 192, 0);
    const v16h a1 = ldsA(&sComb[0][pos * 64], 192, 32);
#pragma unroll
    for (int nt = 0; nt < 16; ++nt) {
      const int n0 = nt * 16;
      v8f acc = zero8();
      acc = wmma_f16(a0, PACKED ? pB(wpk, TB_WIN + nt)
                                : gB(W_in, 256, 0, n0, 64, 256), acc);
      acc = wmma_f16(a1, PACKED ? pB(wpk, TB_WIN + 16 + nt)
                                : gB(W_in, 256, 32, n0, 64, 256), acc);
      const int n = n0 + (lane & 15);
      const int mb = (lane >> 4) << 3;
      if (n0 < 128) {
#pragma unroll
        for (int r = 0; r < 8; ++r) sxc[pos][mb + r][n] = (_Float16)acc[r];
      } else {
#pragma unroll
        for (int r = 0; r < 8; ++r)
          sz[pos][mb + r][n - 128] = (_Float16)siluf(acc[r]);
      }
    }
  }
  __syncthreads();

  // ---- Phase 3: depthwise causal conv (L=2 -> taps 2,3) + SiLU ----
#pragma unroll
  for (int j = 0; j < 4; ++j) {
    const int d = lane + 32 * j;
    const float c2 = sCw2[d], c3 = sCw3[d], cb = sCb[d];
#pragma unroll
    for (int r = 0; r < TM; ++r) {
      const float x0 = (float)sxc[0][r][d];
      const float x1 = (float)sxc[1][r][d];
      sxc[0][r][d] = (_Float16)siluf(x0 * c3 + cb);
      sxc[1][r][d] = (_Float16)siluf(x0 * c2 + x1 * c3 + cb);
    }
  }
  __syncthreads();

  // ---- Phase 4: x_dbl = xc @ W_xproj ; dt / B / C extraction ----
#pragma unroll
  for (int pos = 0; pos < 2; ++pos) {
    const v16h a0 = ldsA(&sxc[pos][0][0], 128, 0);
    const v16h a1 = ldsA(&sxc[pos][0][0], 128, 32);
    const v16h a2 = ldsA(&sxc[pos][0][0], 128, 64);
    const v16h a3 = ldsA(&sxc[pos][0][0], 128, 96);
#pragma unroll
    for (int nt = 0; nt < 3; ++nt) {
      const int n0 = nt * 16;
      v8f acc = zero8();
      acc = wmma_f16(a0, PACKED ? pB(wpk, TB_XPROJ + 0 * 3 + nt)
                                : gB(W_xproj, 36, 0, n0, 128, 36), acc);
      acc = wmma_f16(a1, PACKED ? pB(wpk, TB_XPROJ + 1 * 3 + nt)
                                : gB(W_xproj, 36, 32, n0, 128, 36), acc);
      acc = wmma_f16(a2, PACKED ? pB(wpk, TB_XPROJ + 2 * 3 + nt)
                                : gB(W_xproj, 36, 64, n0, 128, 36), acc);
      acc = wmma_f16(a3, PACKED ? pB(wpk, TB_XPROJ + 3 * 3 + nt)
                                : gB(W_xproj, 36, 96, n0, 128, 36), acc);
      const int n = n0 + (lane & 15);
      const int mb = (lane >> 4) << 3;
#pragma unroll
      for (int r = 0; r < 8; ++r) sXdbl[mb + r][n] = acc[r];
    }
    __syncthreads();
    // dt = softplus(x_dbl[:, :4] @ W_dt + b_dt)  (K=4 -> scalar)
#pragma unroll
    for (int j = 0; j < 4; ++j) {
      const int d = lane + 32 * j;
      const float w0 = sWdt[0][d], w1 = sWdt[1][d];
      const float w2 = sWdt[2][d], w3 = sWdt[3][d];
      const float bd = b_dt[d];
#pragma unroll
      for (int r = 0; r < TM; ++r) {
        const float v = bd + sXdbl[r][0] * w0 + sXdbl[r][1] * w1 +
                        sXdbl[r][2] * w2 + sXdbl[r][3] * w3;
        sdt[pos][r][d] = (_Float16)((v > 20.f) ? v : log1pf(__expf(v)));
      }
    }
    // per-row scan scalars: s00 = B0.C0 ; P = B0*C1 ; s11 = B1.C1
    if (lane < 16) {
      const int r = lane;
      float s = 0.f;
      if (pos == 0) {
#pragma unroll
        for (int nn = 0; nn < 16; ++nn) {
          const float Bv = sXdbl[r][4 + nn];
          const float Cv = sXdbl[r][20 + nn];
          sB0[r][nn] = Bv;
          s += Bv * Cv;
        }
        sS0[r] = s;
      } else {
#pragma unroll
        for (int nn = 0; nn < 16; ++nn) {
          const float Bv = sXdbl[r][4 + nn];
          const float Cv = sXdbl[r][20 + nn];
          sP[r][nn] = sB0[r][nn] * Cv;
          s += Bv * Cv;
        }
        sS1[r] = s;
      }
    }
    __syncthreads();
  }

  // ---- Phase 5: closed-form L=2 selective scan + gate; fold 0.5*mean ----
#pragma unroll
  for (int j = 0; j < 4; ++j) {
    const int d = lane + 32 * j;
    float an[16];
#pragma unroll
    for (int nn = 0; nn < 16; ++nn) an[nn] = sAe[d][nn];
    const float Dd = sDsk[d];
#pragma unroll 4
    for (int r = 0; r < TM; ++r) {
      const float xc0 = (float)sxc[0][r][d];
      const float xc1 = (float)sxc[1][r][d];
      const float dt0 = (float)sdt[0][r][d];
      const float dt1 = (float)sdt[1][r][d];
      const float dtx0 = dt0 * xc0;
      const float dtx1 = dt1 * xc1;
      float S = 0.f;
#pragma unroll
      for (int nn = 0; nn < 16; ++nn) S += __expf(dt1 * an[nn]) * sP[r][nn];
      const float y0 = dtx0 * sS0[r];
      const float y1 = dtx0 * S + dtx1 * sS1[r];
      const float g0 = (y0 + Dd * xc0) * (float)sz[0][r][d];
      const float g1 = (y1 + Dd * xc1) * (float)sz[1][r][d];
      sG[r][d] = (_Float16)(0.5f * (g0 + g1));
    }
  }
  __syncthreads();

  // ---- Phase 6: team_features = (0.5*(g0+g1)) @ W_mout -> global ----
  {
    const v16h a0 = ldsA(&sG[0][0], 128, 0);
    const v16h a1 = ldsA(&sG[0][0], 128, 32);
    const v16h a2 = ldsA(&sG[0][0], 128, 64);
    const v16h a3 = ldsA(&sG[0][0], 128, 96);
    float* __restrict__ out_tf = out + (size_t)3 * BROWS;
#pragma unroll
    for (int nt = 0; nt < 4; ++nt) {
      const int n0 = nt * 16;
      v8f acc = zero8();
      acc = wmma_f16(a0, PACKED ? pB(wpk, TB_MOUT + 0 * 4 + nt)
                                : gB(W_mout, 64, 0, n0, 128, 64), acc);
      acc = wmma_f16(a1, PACKED ? pB(wpk, TB_MOUT + 1 * 4 + nt)
                                : gB(W_mout, 64, 32, n0, 128, 64), acc);
      acc = wmma_f16(a2, PACKED ? pB(wpk, TB_MOUT + 2 * 4 + nt)
                                : gB(W_mout, 64, 64, n0, 128, 64), acc);
      acc = wmma_f16(a3, PACKED ? pB(wpk, TB_MOUT + 3 * 4 + nt)
                                : gB(W_mout, 64, 96, n0, 128, 64), acc);
      const int n = n0 + (lane & 15);
      const int mb = (lane >> 4) << 3;
#pragma unroll
      for (int r = 0; r < 8; ++r)
        __builtin_nontemporal_store(acc[r],
                                    &out_tf[(size_t)(r0 + mb + r) * 64 + n]);
    }
  }

  // ---- Phase 7: h1 = relu(bn2(combined @ W_c1 + b_c1)), K = 192 ----
  {
    v16h a[6];
#pragma unroll
    for (int kt = 0; kt < 6; ++kt) a[kt] = ldsA(&sComb[0][0], 192, kt * 32);
#pragma unroll
    for (int nt = 0; nt < 4; ++nt) {
      const int n0 = nt * 16;
      v8f acc = zero8();
#pragma unroll
      for (int kt = 0; kt < 6; ++kt)
        acc = wmma_f16(a[kt], PACKED ? pB(wpk, TB_C1 + kt * 4 + nt)
                                     : gB(W_c1, 64, kt * 32, n0, 192, 64),
                       acc);
      const int n = n0 + (lane & 15);
      const int mb = (lane >> 4) << 3;
      const float bc = b_c1[n];
      const float g2 = bn2_g[n], bb2 = bn2_b[n], m2 = bn2_m[n];
      const float iv2 = rsqrtf(bn2_v[n] + 1e-5f);
#pragma unroll
      for (int r = 0; r < 8; ++r) {
        const float v = (acc[r] + bc - m2) * iv2 * g2 + bb2;
        sH1[mb + r][n] = v > 0.f ? v : 0.f;
      }
    }
  }
  __syncthreads();

  // ---- Phase 8: logits = h1 @ W_c2 + b_c2 (N=3 -> scalar, lanes 0..15) ----
  if (lane < 16) {
    const int r = lane;
#pragma unroll
    for (int c = 0; c < 3; ++c) {
      float acc = b_c2[c];
#pragma unroll 8
      for (int k = 0; k < 64; ++k) acc += sH1[r][k] * W_c2[k * 3 + c];
      __builtin_nontemporal_store(acc, &out[(size_t)(r0 + r) * 3 + c]);
    }
  }
}

extern "C" void kernel_launch(void* const* d_in, const int* in_sizes, int n_in,
                              void* d_out, int out_size, void* d_ws,
                              size_t ws_size, hipStream_t stream) {
  (void)in_sizes; (void)n_in; (void)out_size;
  const float* x      = (const float*)d_in[0];
  const float* W_home = (const float*)d_in[1];
  const float* b_home = (const float*)d_in[2];
  const float* W_away = (const float*)d_in[3];
  const float* b_away = (const float*)d_in[4];
  const float* W_in   = (const float*)d_in[5];
  const float* conv_w = (const float*)d_in[6];
  const float* conv_b = (const float*)d_in[7];
  const float* W_xproj= (const float*)d_in[8];
  const float* W_dt   = (const float*)d_in[9];
  const float* b_dt   = (const float*)d_in[10];
  const float* A_log  = (const float*)d_in[11];
  const float* D_skip = (const float*)d_in[12];
  const float* W_mout = (const float*)d_in[13];
  const float* W_mc   = (const float*)d_in[14];
  const float* b_mc   = (const float*)d_in[15];
  const float* bn1_g  = (const float*)d_in[16];
  const float* bn1_b  = (const float*)d_in[17];
  const float* bn1_m  = (const float*)d_in[18];
  const float* bn1_v  = (const float*)d_in[19];
  const float* W_c1   = (const float*)d_in[20];
  const float* b_c1   = (const float*)d_in[21];
  const float* bn2_g  = (const float*)d_in[22];
  const float* bn2_b  = (const float*)d_in[23];
  const float* bn2_m  = (const float*)d_in[24];
  const float* bn2_v  = (const float*)d_in[25];
  const float* W_c2   = (const float*)d_in[26];
  const float* b_c2   = (const float*)d_in[27];
  float* out = (float*)d_out;
  _Float16* wpk = (_Float16*)d_ws;

  const bool use_packed =
      (d_ws != nullptr) && (ws_size >= PACK_HALVES * sizeof(_Float16));

  dim3 grid(BROWS / TM), block(32);
  if (use_packed) {
    pack_weights<<<dim3((PACK_HALVES + 255) / 256), dim3(256), 0, stream>>>(
        W_home, W_away, W_mc, W_in, W_xproj, W_mout, W_c1, wpk);
    match_predictor_fused<true><<<grid, block, 0, stream>>>(
        x, W_home, b_home, W_away, b_away, W_in, conv_w, conv_b, W_xproj, W_dt,
        b_dt, A_log, D_skip, W_mout, W_mc, b_mc, bn1_g, bn1_b, bn1_m, bn1_v,
        W_c1, b_c1, bn2_g, bn2_b, bn2_m, bn2_v, W_c2, b_c2, wpk, out);
  } else {
    match_predictor_fused<false><<<grid, block, 0, stream>>>(
        x, W_home, b_home, W_away, b_away, W_in, conv_w, conv_b, W_xproj, W_dt,
        b_dt, A_log, D_skip, W_mout, W_mc, b_mc, bn1_g, bn1_b, bn1_m, bn1_v,
        W_c1, b_c1, bn2_g, bn2_b, bn2_m, bn2_v, W_c2, b_c2, wpk, out);
  }
}
